// RN_1563368096087
// MI455X (gfx1250) — matrix operand order repaired and checked by reading, not compiled, not run on hardware
//
#include <hip/hip_runtime.h>
#include <stdint.h>

typedef __attribute__((ext_vector_type(16))) __bf16 v16bf;
typedef __attribute__((ext_vector_type(8)))  float  v8f;
typedef unsigned short u16;

#define BB 16
#define KK 36
#define CC 2048
#define NPAIR (KK*KK)          // 1296
#define MT 32                  // pair rows per block tile
#define NT_TILES ((NPAIR + MT - 1) / MT)   // 41
#define LDH 1032               // padded LDS row (bf16 elems): 2064B = 129*16B, %64 banks = 4

union Frag16 { uint32_t u[8]; uint4 q[2]; v16bf v; };

__device__ inline u16 f2bf(float f) {
    unsigned u = __float_as_uint(f);
    u += 0x7FFFu + ((u >> 16) & 1u);           // round-to-nearest-even
    return (u16)(u >> 16);
}

// ---------------- prep kernels (one-off, scalar) ----------------

__global__ void k_f2bf(const float* __restrict__ src, u16* __restrict__ dst, int n) {
    int g = blockIdx.x * 256 + threadIdx.x;
    if (g < n) dst[g] = f2bf(src[g]);
}

// W1ab_t[n][k] (2048x2048, n-major): n<1024 -> W1[k][n] ; n>=1024 -> W1[2048+k][n-1024]
__global__ void k_w1abt(const float* __restrict__ W1, u16* __restrict__ dst) {
    int g = blockIdx.x * 256 + threadIdx.x;      // 2048*2048
    int n = g >> 11, k = g & 2047;
    float v = (n < 1024) ? W1[(size_t)k * 1024 + n]
                         : W1[(size_t)(2048 + k) * 1024 + (n - 1024)];
    dst[g] = f2bf(v);
}

// Wt[n][k] = W[k][n], 1024x1024, bf16
__global__ void k_wt(const float* __restrict__ W, u16* __restrict__ dst) {
    int g = blockIdx.x * 256 + threadIdx.x;      // 1024*1024
    int n = g >> 10, k = g & 1023;
    dst[g] = f2bf(W[(size_t)k * 1024 + n]);
}

// Cq[b][n] = q_feats[b] @ W1[4096:5120, n] + b1[n]   (f32, exact)
__global__ void k_cq(const float* __restrict__ q, const float* __restrict__ W1,
                     const float* __restrict__ b1, float* __restrict__ Cq) {
    int g = blockIdx.x * 256 + threadIdx.x;      // 16*1024
    int b = g >> 10, n = g & 1023;
    float s = b1[n];
    for (int k = 0; k < 1024; ++k)
        s += q[b * 1024 + k] * W1[(size_t)(4096 + k) * 1024 + n];
    Cq[g] = s;
}

// dotm[b][p] = box_feats[b,i] . box_feats[b,j]
__global__ void k_dot(const float* __restrict__ bf, float* __restrict__ dotm) {
    int g = blockIdx.x * 256 + threadIdx.x;
    if (g >= BB * NPAIR) return;
    int b = g / NPAIR, p = g % NPAIR;
    int i = p / KK, j = p % KK;
    const float* xi = bf + ((size_t)b * KK + i) * CC;
    const float* xj = bf + ((size_t)b * KK + j) * CC;
    float s = 0.f;
    for (int c = 0; c < CC; ++c) s += xi[c] * xj[c];
    dotm[g] = s;
}

__global__ void k_zero(float* __restrict__ p, int n) {
    int g = blockIdx.x * 256 + threadIdx.x;
    if (g < n) p[g] = 0.f;
}

// ---------------- WMMA GEMM: ABt = box_bf16 (576x2048) @ W1ab_t^T -> f32 576x2048 ----------------

__global__ __launch_bounds__(256) void k_abt(const u16* __restrict__ Abf,
                                             const u16* __restrict__ Bt,
                                             float* __restrict__ out) {
    int mt = blockIdx.x;                 // 0..35
    int nb = blockIdx.y;                 // 0..15
    int w  = threadIdx.x >> 5;
    int l  = threadIdx.x & 31;
    int half = l >> 4, lm = l & 15;
    int n   = nb * 128 + w * 16 + lm;    // output column (N-row of Bt)
    int row = mt * 16 + lm;              // A-matrix row for this lane
    const u16* arow = Abf + (size_t)row * 2048;
    const u16* brow = Bt  + (size_t)n   * 2048;
    v8f acc = {};
    for (int kb = 0; kb < 2048; kb += 32) {
        Frag16 a, b;
        a.q[0] = *(const uint4*)(arow + kb + half * 8);
        a.q[1] = *(const uint4*)(arow + kb + 16 + half * 8);
        b.q[0] = *(const uint4*)(brow + kb + half * 8);
        b.q[1] = *(const uint4*)(brow + kb + 16 + half * 8);
        acc = __builtin_amdgcn_wmma_f32_16x16x32_bf16(false, a.v, false, b.v,
                                                      (short)0, acc, false, false);
    }
#pragma unroll
    for (int r = 0; r < 8; ++r)
        out[(size_t)(mt * 16 + r + 8 * half) * 2048 + n] = acc[r];
}

// ---------------- fused main kernel: h1 -> h2 -> h3 -> masked column sum ----------------

__global__ __launch_bounds__(256) void k_main(
    const float* __restrict__ ABt,    // 576 x 2048 : [A | Bt]
    const float* __restrict__ Cq,     // 16 x 1024 (includes b1)
    const float* __restrict__ dotm,   // 16 x 1296
    const float* __restrict__ W1,     // 5136 x 1024 (geom rows 5120..5135 used)
    const float* __restrict__ bcoord, // 16 x 36 x 6
    const int*   __restrict__ indexv, // 16
    const u16*   __restrict__ W2t,    // 1024 x 1024 n-major bf16
    const float* __restrict__ b2,
    const u16*   __restrict__ W3t,
    const float* __restrict__ b3,
    float* __restrict__ g)            // 16 x 1024 accumulators (pre-zeroed)
{
    __shared__ __align__(16) u16 sh1[MT * LDH];
    __shared__ __align__(16) u16 sh2[MT * LDH];
    __shared__ float geomf[MT][16];
    __shared__ float maskv[MT];

    const int b    = blockIdx.y;
    const int tile = blockIdx.x;
    const int tid  = threadIdx.x;
    const int w    = tid >> 5;
    const int l    = tid & 31;
    const int half = l >> 4, lm = l & 15;

    // ---- phase 0: per-row geometry features + pair mask ----
    if (tid < MT) {
        int m = tid;
        int p = tile * MT + m;
        int pc = p < NPAIR ? p : NPAIR - 1;
        int i = pc / KK, j = pc % KK;
        const float* bi = bcoord + ((size_t)b * KK + i) * 6;
        const float* bj = bcoord + ((size_t)b * KK + j) * 6;
        float ai = (bi[2] - bi[0]) * (bi[3] - bi[1]);
        float aj = (bj[2] - bj[0]) * (bj[3] - bj[1]);
        float lr = fmaxf(0.f, fminf(bi[2], bj[2]) - fmaxf(bi[0], bj[0]));
        float ud = fmaxf(0.f, fminf(bi[3], bj[3]) - fmaxf(bi[1], bj[1]));
        float ov = lr * ud;
        geomf[m][0] = dotm[(size_t)b * NPAIR + pc];
#pragma unroll
        for (int t = 0; t < 6; ++t) { geomf[m][1 + t] = bi[t]; geomf[m][7 + t] = bj[t]; }
        geomf[m][13] = ov / (ai + aj - ov);
        geomf[m][14] = ov / ai;
        geomf[m][15] = ov / aj;
        int idx = indexv[b];
        maskv[m] = (p < NPAIR && i < idx && j < idx) ? 1.f : 0.f;
    }
    __syncthreads();

    // ---- phase 1: h1 = relu(A[i] + Bt[j] + Cq + geom@W1g) -> LDS bf16 ----
    // Each thread owns 4 fixed columns n = tid + 256*q; hoist W1g/Cq per column
    // into registers (2 columns at a time) and reuse across all 32 rows.
    for (int qc = 0; qc < 2; ++qc) {
        int na = tid + 512 * qc;
        int nb2 = na + 256;
        float cqa = Cq[(size_t)b * 1024 + na];
        float cqb = Cq[(size_t)b * 1024 + nb2];
        float w1ga[16], w1gb[16];
#pragma unroll
        for (int t = 0; t < 16; ++t) {
            w1ga[t] = W1[(size_t)(5120 + t) * 1024 + na];
            w1gb[t] = W1[(size_t)(5120 + t) * 1024 + nb2];
        }
        for (int m = 0; m < MT; ++m) {
            int p = tile * MT + m;
            int pc = p < NPAIR ? p : NPAIR - 1;
            int i = pc / KK, j = pc % KK;
            const float* ra = ABt + ((size_t)b * KK + i) * 2048;
            const float* rb = ABt + ((size_t)b * KK + j) * 2048 + 1024;
            float va = ra[na] + rb[na] + cqa;
            float vb = ra[nb2] + rb[nb2] + cqb;
#pragma unroll
            for (int t = 0; t < 16; ++t) {
                float gm = geomf[m][t];
                va += gm * w1ga[t];
                vb += gm * w1gb[t];
            }
            sh1[m * LDH + na]  = f2bf(fmaxf(va, 0.f));
            sh1[m * LDH + nb2] = f2bf(fmaxf(vb, 0.f));
        }
    }
    __syncthreads();

    // ---- phase 2: h2 = relu(h1 @ W2 + b2); 2 M-subtiles x 2 N-tiles per step ----
    {
        const u16* a0 = sh1 + lm * LDH;
        const u16* a1 = sh1 + (16 + lm) * LDH;
        for (int s = 0; s < 4; ++s) {
            int n0 = (w * 8 + s * 2) * 16 + lm;
            int n1 = n0 + 16;
            const u16* br0 = W2t + (size_t)n0 * 1024;
            const u16* br1 = W2t + (size_t)n1 * 1024;
            v8f acc00 = {}, acc10 = {}, acc01 = {}, acc11 = {};
#pragma unroll 2
            for (int kb = 0; kb < 1024; kb += 32) {
                Frag16 fa0, fa1, fb0, fb1;
                int ko = kb + half * 8;
                fa0.q[0] = *(const uint4*)(a0 + ko);
                fa0.q[1] = *(const uint4*)(a0 + ko + 16);
                fa1.q[0] = *(const uint4*)(a1 + ko);
                fa1.q[1] = *(const uint4*)(a1 + ko + 16);
                int kbo = kb + half * 8;
                fb0.q[0] = *(const uint4*)(br0 + kbo);
                fb0.q[1] = *(const uint4*)(br0 + kbo + 16);
                fb1.q[0] = *(const uint4*)(br1 + kbo);
                fb1.q[1] = *(const uint4*)(br1 + kbo + 16);
                acc00 = __builtin_amdgcn_wmma_f32_16x16x32_bf16(false, fa0.v, false, fb0.v,
                                                                (short)0, acc00, false, false);
                acc10 = __builtin_amdgcn_wmma_f32_16x16x32_bf16(false, fa1.v, false, fb0.v,
                                                                (short)0, acc10, false, false);
                acc01 = __builtin_amdgcn_wmma_f32_16x16x32_bf16(false, fa0.v, false, fb1.v,
                                                                (short)0, acc01, false, false);
                acc11 = __builtin_amdgcn_wmma_f32_16x16x32_bf16(false, fa1.v, false, fb1.v,
                                                                (short)0, acc11, false, false);
            }
            float bv0 = b2[n0], bv1 = b2[n1];
#pragma unroll
            for (int r = 0; r < 8; ++r) {
                int r0 = r + 8 * half;
                sh2[r0 * LDH + n0]        = f2bf(fmaxf(acc00[r] + bv0, 0.f));
                sh2[(16 + r0) * LDH + n0] = f2bf(fmaxf(acc10[r] + bv0, 0.f));
                sh2[r0 * LDH + n1]        = f2bf(fmaxf(acc01[r] + bv1, 0.f));
                sh2[(16 + r0) * LDH + n1] = f2bf(fmaxf(acc11[r] + bv1, 0.f));
            }
        }
    }
    __syncthreads();

    // ---- phase 3: h3 = relu(h2 @ W3 + b3); masked column-sum -> atomicAdd g[b] ----
    {
        const u16* a0 = sh2 + lm * LDH;
        const u16* a1 = sh2 + (16 + lm) * LDH;
        for (int s = 0; s < 4; ++s) {
            int n0 = (w * 8 + s * 2) * 16 + lm;
            int n1 = n0 + 16;
            const u16* br0 = W3t + (size_t)n0 * 1024;
            const u16* br1 = W3t + (size_t)n1 * 1024;
            v8f acc00 = {}, acc10 = {}, acc01 = {}, acc11 = {};
#pragma unroll 2
            for (int kb = 0; kb < 1024; kb += 32) {
                Frag16 fa0, fa1, fb0, fb1;
                int ko = kb + half * 8;
                fa0.q[0] = *(const uint4*)(a0 + ko);
                fa0.q[1] = *(const uint4*)(a0 + ko + 16);
                fa1.q[0] = *(const uint4*)(a1 + ko);
                fa1.q[1] = *(const uint4*)(a1 + ko + 16);
                int kbo = kb + half * 8;
                fb0.q[0] = *(const uint4*)(br0 + kbo);
                fb0.q[1] = *(const uint4*)(br0 + kbo + 16);
                fb1.q[0] = *(const uint4*)(br1 + kbo);
                fb1.q[1] = *(const uint4*)(br1 + kbo + 16);
                acc00 = __builtin_amdgcn_wmma_f32_16x16x32_bf16(false, fa0.v, false, fb0.v,
                                                                (short)0, acc00, false, false);
                acc10 = __builtin_amdgcn_wmma_f32_16x16x32_bf16(false, fa1.v, false, fb0.v,
                                                                (short)0, acc10, false, false);
                acc01 = __builtin_amdgcn_wmma_f32_16x16x32_bf16(false, fa0.v, false, fb1.v,
                                                                (short)0, acc01, false, false);
                acc11 = __builtin_amdgcn_wmma_f32_16x16x32_bf16(false, fa1.v, false, fb1.v,
                                                                (short)0, acc11, false, false);
            }
            float bv0 = b3[n0], bv1 = b3[n1];
            float s0 = 0.f, s1 = 0.f;
#pragma unroll
            for (int r = 0; r < 8; ++r) {
                int r0 = r + 8 * half;
                float m0 = maskv[r0], m1 = maskv[16 + r0];
                s0 += fmaxf(acc00[r] + bv0, 0.f) * m0;
                s0 += fmaxf(acc10[r] + bv0, 0.f) * m1;
                s1 += fmaxf(acc01[r] + bv1, 0.f) * m0;
                s1 += fmaxf(acc11[r] + bv1, 0.f) * m1;
            }
            atomicAdd(&g[(size_t)b * 1024 + n0], s0);
            atomicAdd(&g[(size_t)b * 1024 + n1], s1);
        }
    }
}

// ---------------- tiny final MLP (f32 scalar, negligible) ----------------

__global__ void k_g1(const float* __restrict__ g, const float* __restrict__ Wf1,
                     const float* __restrict__ bf1, float* __restrict__ g1) {
    int t = blockIdx.x * 256 + threadIdx.x;      // 16*512
    int b = t >> 9, c = t & 511;
    float s = bf1[c];
    for (int k = 0; k < 1024; ++k) s += g[b * 1024 + k] * Wf1[(size_t)k * 512 + c];
    g1[t] = fmaxf(s, 0.f);
}

__global__ void k_out(const float* __restrict__ g1, const float* __restrict__ Wf2,
                      const float* __restrict__ bf2, float* __restrict__ out) {
    int t = threadIdx.x;                         // 256 = 16*16
    int b = t >> 4, c = t & 15;
    float s = bf2[c];
    for (int k = 0; k < 512; ++k) s += g1[b * 512 + k] * Wf2[(size_t)k * 16 + c];
    out[t] = s;
}

// ---------------- host launcher ----------------

extern "C" void kernel_launch(void* const* d_in, const int* in_sizes, int n_in,
                              void* d_out, int out_size, void* d_ws, size_t ws_size,
                              hipStream_t stream) {
    const float* box_feats = (const float*)d_in[2];   // 16x36x2048
    const float* q_feats   = (const float*)d_in[3];   // 16x1024
    const float* bcoord    = (const float*)d_in[4];   // 16x36x6
    const int*   indexv    = (const int*)  d_in[5];   // 16
    const float* W1 = (const float*)d_in[6];
    const float* b1 = (const float*)d_in[7];
    const float* W2 = (const float*)d_in[8];
    const float* b2 = (const float*)d_in[9];
    const float* W3 = (const float*)d_in[10];
    const float* b3 = (const float*)d_in[11];
    const float* Wf1 = (const float*)d_in[12];
    const float* bf1 = (const float*)d_in[13];
    const float* Wf2 = (const float*)d_in[14];
    const float* bf2 = (const float*)d_in[15];
    float* out = (float*)d_out;

    char* ws = (char*)d_ws;
    size_t off = 0;
    auto alloc = [&](size_t bytes) -> void* {
        off = (off + 255) & ~(size_t)255;
        void* p = ws + off;
        off += bytes;
        return p;
    };
    u16*   box_bf = (u16*)  alloc((size_t)576 * 2048 * 2);
    u16*   W1abt  = (u16*)  alloc((size_t)2048 * 2048 * 2);
    u16*   W2t    = (u16*)  alloc((size_t)1024 * 1024 * 2);
    u16*   W3t    = (u16*)  alloc((size_t)1024 * 1024 * 2);
    float* ABt    = (float*)alloc((size_t)576 * 2048 * 4);
    float* Cq     = (float*)alloc((size_t)16 * 1024 * 4);
    float* dotm   = (float*)alloc((size_t)BB * NPAIR * 4);
    float* gacc   = (float*)alloc((size_t)16 * 1024 * 4);
    float* g1     = (float*)alloc((size_t)16 * 512 * 4);

    // prep
    k_f2bf <<<(576 * 2048 + 255) / 256, 256, 0, stream>>>(box_feats, box_bf, 576 * 2048);
    k_w1abt<<<(2048 * 2048) / 256,      256, 0, stream>>>(W1, W1abt);
    k_wt   <<<(1024 * 1024) / 256,      256, 0, stream>>>(W2, W2t);
    k_wt   <<<(1024 * 1024) / 256,      256, 0, stream>>>(W3, W3t);
    k_cq   <<<(16 * 1024) / 256,        256, 0, stream>>>(q_feats, W1, b1, Cq);
    k_dot  <<<(BB * NPAIR + 255) / 256, 256, 0, stream>>>(box_feats, dotm);
    k_zero <<<(16 * 1024) / 256,        256, 0, stream>>>(gacc, 16 * 1024);

    // factorized layer-1 GEMM: ABt = box @ [W1a|W1b]
    k_abt<<<dim3(36, 16), 256, 0, stream>>>(box_bf, W1abt, ABt);

    // fused layers 1..3 + masked reduction
    k_main<<<dim3(NT_TILES, BB), 256, 0, stream>>>(ABt, Cq, dotm, W1, bcoord, indexv,
                                                   W2t, b2, W3t, b3, gacc);

    // final MLP
    k_g1 <<<(16 * 512) / 256, 256, 0, stream>>>(gacc, Wf1, bf1, g1);
    k_out<<<1, 256, 0, stream>>>(g1, Wf2, bf2, out);
}
